// Dblock_ViT_28381143892136
// MI455X (gfx1250) — matrix operand order repaired and checked by reading, not compiled, not run on hardware
//
#include <hip/hip_runtime.h>

typedef __bf16 bf16_t;
typedef bf16_t v2bf  __attribute__((ext_vector_type(2)));
typedef bf16_t v8bf  __attribute__((ext_vector_type(8)));
typedef bf16_t v16bf __attribute__((ext_vector_type(16)));
typedef float  v8f   __attribute__((ext_vector_type(8)));

__device__ __forceinline__ float ldf(const float* p)  { return *p; }
__device__ __forceinline__ float ldf(const bf16_t* p) { return (float)*p; }

__device__ __forceinline__ unsigned pack_bf16(float lo, float hi)
{
    v2bf p;
    p[0] = (bf16_t)lo;
    p[1] = (bf16_t)hi;
    unsigned u;
    __builtin_memcpy(&u, &p, 4);
    return u;
}

__device__ __forceinline__ v16bf cat8(v8bf a, v8bf b)
{
    return __builtin_shufflevector(a, b, 0, 1, 2, 3, 4, 5, 6, 7,
                                         8, 9, 10, 11, 12, 13, 14, 15);
}

// ---------------------------------------------------------------------------
// Generic batched NT GEMM:  D[m,n] = alpha * sum_k A(m,k) * B(n,k) (+ bias[n])
// REQUIRES: M,N multiples of 64; K and each split-K chunk multiples of 32
// (all launches below satisfy this). A tile staged [m][k], B tile staged
// TRANSPOSED [k][n]. Each A fragment is two contiguous 16-byte LDS
// loads (ds_load_b128); B: 16 strided bf16 loads per fragment; no cross-lane assembly. Compute via
// v_wmma_f32_16x16x32_bf16. kSplit>1 -> split-K with atomicAdd (D pre-zeroed).
// ---------------------------------------------------------------------------
template<typename TA, typename TB>
__global__ __launch_bounds__(128)
void gemm_nt(const TA* __restrict__ A, long sAm, long sAk, long sAb,
             const TB* __restrict__ B, long sBn, long sBk, long sBb,
             float* __restrict__ D, long sDm, long sDb,
             const float* __restrict__ bias,
             int M, int N, int K, float alpha, int kSplit)
{
    // A: 64 rows x 32 bf16, row stride 20 uints (80B, 16B-aligned, bank-spread)
    // Bt: 32 k-rows x 64 bf16, row stride 36 uints (144B, 16B-aligned)
    __shared__ unsigned As32[64][20];
    __shared__ unsigned Bt32[32][36];
    const bf16_t (*As)[40] = (const bf16_t (*)[40])As32;
    const bf16_t (*Bt)[72] = (const bf16_t (*)[72])Bt32;

    const int tid  = threadIdx.x;
    const int lane = tid & 31;
    const int wave = tid >> 5;
    const int wm   = (wave >> 1) * 32;   // wave tile origin in block tile
    const int wn   = (wave & 1) * 32;

    const int bm    = blockIdx.x * 64;
    const int bn    = blockIdx.y * 64;
    const int batch = blockIdx.z / kSplit;
    const int slice = blockIdx.z % kSplit;
    int kLen = ((K + kSplit - 1) / kSplit + 31) & ~31;
    const int kBeg = slice * kLen;
    const int kEnd = (kBeg + kLen < K) ? (kBeg + kLen) : K;

    A += (long)batch * sAb;
    B += (long)batch * sBb;
    D += (long)batch * sDb;

    // A fill: thread owns k-pair (tid&15); rows (tid>>4) + 8*i
    const int kp  = tid & 15;
    const int rwA = tid >> 4;
    const TA* apBase = A + (long)(bm + rwA) * sAm + (long)(kBeg + 2 * kp) * sAk;
    // B fill: thread owns n-pair (tid&31); k rows (tid>>5) + 4*i
    const int np  = tid & 31;
    const int krB = tid >> 5;
    const TB* bpBase = B + (long)(bn + 2 * np) * sBn + (long)(kBeg + krB) * sBk;

    const long aRowStep = 8 * sAm, bRowStep = 4 * sBk;
    const long aChunk = 32 * sAk, bChunk = 32 * sBk;

    v8f acc[2][2] = {};

    for (int k0 = kBeg; k0 < kEnd; k0 += 32) {
        // ---- stage A tile [m][k] and B tile transposed [k][n] into LDS ----
        {
            const TA* ap = apBase;
            const TB* bp = bpBase;
            #pragma unroll
            for (int i = 0; i < 8; ++i) {
                float a0 = ldf(ap), a1 = ldf(ap + sAk);
                float b0 = ldf(bp), b1 = ldf(bp + sBn);
                As32[rwA + 8 * i][kp] = pack_bf16(a0, a1);
                Bt32[krB + 4 * i][np] = pack_bf16(b0, b1);
                ap += aRowStep;
                bp += bRowStep;
            }
        }
        apBase += aChunk;
        bpBase += bChunk;
        if (k0 + 32 < kEnd) {           // hint next K-chunk toward caches
            __builtin_prefetch(apBase, 0, 1);
            __builtin_prefetch(bpBase, 0, 1);
        }
        __syncthreads();

        // ---- fragments: A = two contiguous 16B LDS loads, B = 16 column loads (ISA layouts) ----
        v16bf afrag[2], bfrag[2];
        const int am = lane & 15;            // A row within 16
        const int kh = (lane >> 4) * 8;      // K half select
        #pragma unroll
        for (int tm = 0; tm < 2; ++tm) {
            const bf16_t* ar = As[wm + tm * 16 + am];
            v8bf c0 = *(const v8bf*)(ar + kh);        // K = kh .. kh+7
            v8bf c1 = *(const v8bf*)(ar + 16 + kh);   // K = 16+kh .. 16+kh+7
            afrag[tm] = cat8(c0, c1);
        }
        #pragma unroll
        for (int tn = 0; tn < 2; ++tn) {
            const int bcol = wn + tn * 16 + am;  // N col = lane&15
            #pragma unroll
            for (int s = 0; s < 16; ++s)
                bfrag[tn][s] = Bt[(s < 8) ? (kh + s) : (16 + kh + s - 8)][bcol];
        }

        #pragma unroll
        for (int tm = 0; tm < 2; ++tm)
            #pragma unroll
            for (int tn = 0; tn < 2; ++tn)
                acc[tm][tn] = __builtin_amdgcn_wmma_f32_16x16x32_bf16(
                    false, afrag[tm], false, bfrag[tn],
                    (short)0, acc[tm][tn], false, false);

        __syncthreads();
    }

    // ---- epilogue: C/D layout VGPR v -> M = v + 8*(lane>>4), N = lane&15 ----
    const int cn = lane & 15;
    const int cm = (lane >> 4) * 8;
    #pragma unroll
    for (int tm = 0; tm < 2; ++tm)
        #pragma unroll
        for (int tn = 0; tn < 2; ++tn) {
            int gn = bn + wn + tn * 16 + cn;
            float badd = (kSplit == 1 && bias) ? bias[gn] : 0.f;
            #pragma unroll
            for (int v = 0; v < 8; ++v) {
                int gm = bm + wm + tm * 16 + cm + v;
                float val = alpha * acc[tm][tn][v];
                if (kSplit > 1)
                    atomicAdd(&D[(long)gm * sDm + gn], val);
                else
                    D[(long)gm * sDm + gn] = val + badd;
            }
        }
    (void)M; (void)N;
}

// ---------------------------------------------------------------------------
// Grouped 3x3 conv, groups = C/2 (2 in-channels per group), SAME zero pad.
// x: [8,128,128,128] f32, w: [128,2,3,3] f32, out: same shape (f32 or bf16).
// ---------------------------------------------------------------------------
template<typename OutT>
__global__ __launch_bounds__(256)
void gconv3x3(const float* __restrict__ x, const float* __restrict__ w,
              OutT* __restrict__ out)
{
    long idx = (long)blockIdx.x * 256 + threadIdx.x;   // 8*128*16384 total
    int px = (int)(idx & 127);
    int py = (int)((idx >> 7) & 127);
    int c  = (int)((idx >> 14) & 127);
    int bi = (int)(idx >> 21);
    const float* wr = w + c * 18;
    const float* x0 = x + (((long)bi * 128 + (c & ~1)) << 14);
    float s = 0.f;
    #pragma unroll
    for (int j = 0; j < 2; ++j) {
        const float* xp = x0 + ((long)j << 14);
        #pragma unroll
        for (int dy = -1; dy <= 1; ++dy) {
            int yy = py + dy;
            if ((unsigned)yy >= 128u) continue;
            #pragma unroll
            for (int dx = -1; dx <= 1; ++dx) {
                int xx = px + dx;
                if ((unsigned)xx >= 128u) continue;
                s += wr[j * 9 + (dy + 1) * 3 + (dx + 1)] * xp[yy * 128 + xx];
            }
        }
    }
    out[idx] = (OutT)s;
}

// ---------------------------------------------------------------------------
// Row-wise L2 normalize (torch F.normalize semantics, eps=1e-12), f32 -> bf16.
// One 256-thread block per row.
// ---------------------------------------------------------------------------
__global__ __launch_bounds__(256)
void l2norm_rows(const float* __restrict__ x, bf16_t* __restrict__ y, int rowLen)
{
    long base = (long)blockIdx.x * rowLen;
    float ss = 0.f;
    for (int i = threadIdx.x; i < rowLen; i += 256) {
        float v = x[base + i];
        ss += v * v;
    }
    #pragma unroll
    for (int m = 16; m >= 1; m >>= 1) ss += __shfl_xor(ss, m, 32);
    __shared__ float red[8];
    __shared__ float scale;
    if ((threadIdx.x & 31) == 0) red[threadIdx.x >> 5] = ss;
    __syncthreads();
    if (threadIdx.x == 0) {
        float t = 0.f;
        for (int i = 0; i < 8; ++i) t += red[i];
        scale = 1.f / fmaxf(sqrtf(t), 1e-12f);
    }
    __syncthreads();
    float sc = scale;
    for (int i = threadIdx.x; i < rowLen; i += 256)
        y[base + i] = (bf16_t)(x[base + i] * sc);
}

// ---------------------------------------------------------------------------
// Per-batch InstanceNorm2d(1) over the 128x128 attn matrix + softmax(axis=e).
// Block per batch (128 threads); thread c owns attn row c.
// ---------------------------------------------------------------------------
__global__ __launch_bounds__(128)
void instnorm_softmax(const float* __restrict__ attn, float* __restrict__ P)
{
    int b = blockIdx.x, c = threadIdx.x;
    const float* row  = attn + ((long)b * 128 + c) * 128;
    float*       prow = P    + ((long)b * 128 + c) * 128;

    float s = 0.f, q = 0.f;
    for (int e = 0; e < 128; ++e) { float v = row[e]; s += v; q += v * v; }

    __shared__ float rs[128], rq[128];
    rs[c] = s; rq[c] = q; __syncthreads();
    for (int off = 64; off >= 1; off >>= 1) {
        if (c < off) { rs[c] += rs[c + off]; rq[c] += rq[c + off]; }
        __syncthreads();
    }
    float mean = rs[0] * (1.f / 16384.f);
    float var  = rq[0] * (1.f / 16384.f) - mean * mean;
    float inv  = rsqrtf(var + 1e-5f);

    float mx = -3.4e38f;
    for (int e = 0; e < 128; ++e) mx = fmaxf(mx, (row[e] - mean) * inv);
    float sum = 0.f;
    for (int e = 0; e < 128; ++e) {
        float t = __expf((row[e] - mean) * inv - mx);
        prow[e] = t;
        sum += t;
    }
    float rcp = 1.f / sum;
    for (int e = 0; e < 128; ++e) prow[e] *= rcp;
}

// ---------------------------------------------------------------------------
extern "C" void kernel_launch(void* const* d_in, const int* in_sizes, int n_in,
                              void* d_out, int out_size, void* d_ws, size_t ws_size,
                              hipStream_t stream)
{
    const float* emb[4] = {(const float*)d_in[0], (const float*)d_in[1],
                           (const float*)d_in[2], (const float*)d_in[3]};
    const float* text = (const float*)d_in[4];   // [8,1,128,512] -> 1024x512
    const float* Wq   = (const float*)d_in[5];   // [16384,512]
    const float* bq   = (const float*)d_in[6];   // [16384]
    const float* Wmk  = (const float*)d_in[7];   // [4,128,128]
    const float* Wk   = (const float*)d_in[8];   // [4,128,2,3,3]
    const float* Wmv  = (const float*)d_in[9];
    const float* Wv   = (const float*)d_in[10];
    const float* Wpo  = (const float*)d_in[11];
    float* out = (float*)d_out;

    const long HW  = 16384;           // 128*128
    const long PB  = 128L * HW;       // per-batch plane stride
    const long TOT = 8L * PB;         // 16,777,216 elements

    char* ws = (char*)d_ws;
    size_t off = 0;
    auto take = [&](size_t bytes) -> char* {
        char* p = ws + off;
        off += (bytes + 255) & ~(size_t)255;
        return p;
    };
    float*  yf   = (float*)take(TOT * 4);      // conv1x1 scratch (also q_f32)
    float*  kkf  = (float*)take(TOT * 4);      // k pre-norm scratch
    bf16_t* qbf  = (bf16_t*)take(TOT * 2);     // normalized q, bf16
    bf16_t* kkbf = (bf16_t*)take(TOT * 2);     // normalized k, bf16
    bf16_t* vvbf = (bf16_t*)take(TOT * 2);     // v, bf16
    float*  attn = (float*)take(8L * 128 * 128 * 4);
    float*  Pbuf = (float*)take(8L * 128 * 128 * 4);
    float*  Mbuf = (float*)take(8L * 128 * 128 * 4);
    float*  qf   = yf;                         // alias: q_f32 lives before branches

    dim3 blk(128);

    // q = L2norm(text @ Wq^T + bq)   (M=1024, N=16384, K=512)
    gemm_nt<float, float><<<dim3(16, 256, 1), blk, 0, stream>>>(
        text, 512, 1, 0,  Wq, 512, 1, 0,
        qf, HW, 0, bq, 1024, 16384, 512, 1.f, 1);
    l2norm_rows<<<1024, 256, 0, stream>>>(qf, qbf, 16384);

    for (int r = 0; r < 4; ++r) {
        // ---- k path: 1x1 conv (GEMM), grouped 3x3 conv, row L2 norm ----
        gemm_nt<float, float><<<dim3(2, 256, 8), blk, 0, stream>>>(
            Wmk + (long)r * 128 * 128, 128, 1, 0,
            emb[r], 1, HW, PB,
            yf, HW, PB, nullptr, 128, 16384, 128, 1.f, 1);
        gconv3x3<float><<<65536, 256, 0, stream>>>(yf, Wk + (long)r * 128 * 18, kkf);
        l2norm_rows<<<1024, 256, 0, stream>>>(kkf, kkbf, 16384);

        // ---- v path: 1x1 conv (GEMM), grouped 3x3 conv -> bf16 ----
        gemm_nt<float, float><<<dim3(2, 256, 8), blk, 0, stream>>>(
            Wmv + (long)r * 128 * 128, 128, 1, 0,
            emb[r], 1, HW, PB,
            yf, HW, PB, nullptr, 128, 16384, 128, 1.f, 1);
        gconv3x3<bf16_t><<<65536, 256, 0, stream>>>(yf, Wv + (long)r * 128 * 18, vvbf);

        // ---- attn = (qn @ kn^T)/sqrt(C): split-K=16 over K=16384 ----
        hipMemsetAsync(attn, 0, (size_t)8 * 128 * 128 * 4, stream);
        gemm_nt<bf16_t, bf16_t><<<dim3(2, 2, 8 * 16), blk, 0, stream>>>(
            qbf, HW, 1, PB,  kkbf, HW, 1, PB,
            attn, 128, 128L * 128, nullptr,
            128, 128, 16384, 0.08838834764831845f, 16);

        // ---- InstanceNorm2d(1) + softmax over e ----
        instnorm_softmax<<<8, 128, 0, stream>>>(attn, Pbuf);

        // ---- M = Wpo @ P  (tiny 128^3 per batch; fuses project_out) ----
        gemm_nt<float, float><<<dim3(2, 2, 8), blk, 0, stream>>>(
            Wpo + (long)r * 128 * 128, 128, 1, 0,
            Pbuf, 1, 128, 128L * 128,
            Mbuf, 128, 128L * 128, nullptr, 128, 128, 128, 1.f, 1);

        // ---- out_r = M @ V, written straight to d_out ----
        gemm_nt<float, bf16_t><<<dim3(2, 256, 8), blk, 0, stream>>>(
            Mbuf, 128, 1, 128L * 128,
            vvbf, 1, HW, PB,
            out + (size_t)r * TOT, HW, PB, nullptr, 128, 16384, 128, 1.f, 1);
    }

    (void)in_sizes; (void)n_in; (void)out_size; (void)ws_size;
}
